// RWKV5_3745211482484
// MI455X (gfx1250) — compile-verified, matrix operand order checked by reading, not run on hardware
//
#include <hip/hip_runtime.h>

// ------------------------------------------------------------------
// RWKV5 forward for MI455X (gfx1250, wave32, WMMA bf16 16x16x32,
// TDM tensor_load_to_lds staging when available)
// ------------------------------------------------------------------
#define T_LEN 4096
#define C_DIM 1024
#define NHEAD 16
#define HSZ   64
#define FFN   3584
#define VOCABN 32000
#define NLAYER 2

typedef __attribute__((ext_vector_type(16))) __bf16 v16bf;
typedef __attribute__((ext_vector_type(8)))  float  v8f;
typedef __attribute__((ext_vector_type(4))) unsigned int u32x4;
typedef __attribute__((ext_vector_type(8))) int i32x8;
typedef __attribute__((ext_vector_type(4))) int i32x4;

#if defined(__AMDGCN__) && __has_builtin(__builtin_amdgcn_tensor_load_to_lds) && \
    __has_builtin(__builtin_amdgcn_s_wait_tensorcnt)
#define USE_TDM 1
#else
#define USE_TDM 0
#endif

// ---------------- embedding gather ----------------
__global__ void k_gather(const float* __restrict__ emb, const int* __restrict__ ids,
                         float* __restrict__ x) {
  int t = blockIdx.x;
  const float* src = emb + (size_t)ids[t] * C_DIM;
  float* dst = x + (size_t)t * C_DIM;
  for (int c = threadIdx.x; c < C_DIM; c += blockDim.x) dst[c] = src[c];
}

// ---------------- LayerNorm over C=1024 (in-place safe) ----------------
__global__ __launch_bounds__(256)
void k_layernorm(const float* __restrict__ in, float* __restrict__ out,
                 const float* __restrict__ w, const float* __restrict__ b) {
  __shared__ float red[256];
  int t = blockIdx.x, tid = threadIdx.x;
  const float* row = in + (size_t)t * C_DIM;
  float v0 = row[tid], v1 = row[tid + 256], v2 = row[tid + 512], v3 = row[tid + 768];
  red[tid] = v0 + v1 + v2 + v3; __syncthreads();
  for (int st = 128; st; st >>= 1) { if (tid < st) red[tid] += red[tid + st]; __syncthreads(); }
  float m = red[0] * (1.f / C_DIM);
  __syncthreads();
  float d0 = v0 - m, d1 = v1 - m, d2 = v2 - m, d3 = v3 - m;
  red[tid] = d0 * d0 + d1 * d1 + d2 * d2 + d3 * d3; __syncthreads();
  for (int st = 128; st; st >>= 1) { if (tid < st) red[tid] += red[tid + st]; __syncthreads(); }
  float rstd = rsqrtf(red[0] * (1.f / C_DIM) + 1e-5f);
  float* orow = out + (size_t)t * C_DIM;
  orow[tid]       = d0 * rstd * w[tid]       + b[tid];
  orow[tid + 256] = d1 * rstd * w[tid + 256] + b[tid + 256];
  orow[tid + 512] = d2 * rstd * w[tid + 512] + b[tid + 512];
  orow[tid + 768] = d3 * rstd * w[tid + 768] + b[tid + 768];
}

// ---------------- token-shift mix -> bf16 ----------------
__global__ void k_mix(const float* __restrict__ cur, const float* __restrict__ p,
                      __bf16* __restrict__ out) {
  int t = blockIdx.x, tid = threadIdx.x;
  const float* row = cur + (size_t)t * C_DIM;
  for (int c = tid; c < C_DIM; c += 256) {
    float pc = p[c];
    float pv = (t > 0) ? row[c - C_DIM] : 0.0f;
    out[(size_t)t * C_DIM + c] = (__bf16)(row[c] * pc + pv * (1.0f - pc));
  }
}

// ---------------- f32 -> bf16 convert (grid-stride) ----------------
__global__ void k_cvt(const float* __restrict__ in, __bf16* __restrict__ out, size_t n) {
  size_t i = (size_t)blockIdx.x * blockDim.x + threadIdx.x;
  size_t stride = (size_t)gridDim.x * blockDim.x;
  for (; i < n; i += stride) out[i] = (__bf16)in[i];
}

// ---------------- WMMA bf16 GEMM: C[M,N] = A[M,K] * B[K,N] ----------------
// 256 threads = 8 waves; block tile 128x128; wave tile 64x32 (4x2 wmma accs).
// Requires M%128==0, N%128==0, K%32==0 (true for all GEMMs here).
// A tile (128x32, [m][k] row-major) is staged by the Tensor Data Mover when
// the toolchain exposes tensor_load_to_lds; B tile is staged with a register
// transpose to [n][k] so fragments are two b128 LDS loads.
__global__ __launch_bounds__(256)
void k_gemm_bf16(const __bf16* __restrict__ A, const __bf16* __restrict__ Bm,
                 float* __restrict__ Cm, int M, int N, int K) {
  __shared__ __bf16 lA[128][32];   // [m][k]
  __shared__ __bf16 lB[128][32];   // [n][k] (transposed on stage-in)
  int tid = threadIdx.x;
  int m0 = blockIdx.y * 128, n0 = blockIdx.x * 128;
  int wid = tid >> 5, lane = tid & 31;
  int mBase = (wid & 1) * 64, nBase = (wid >> 1) * 32;
  int lm = lane & 15, lh = lane >> 4;
  int kb = lh * 8;

  const v8f vzero = {0.f, 0.f, 0.f, 0.f, 0.f, 0.f, 0.f, 0.f};
  v8f acc[4][2];
#pragma unroll
  for (int i = 0; i < 4; i++)
#pragma unroll
    for (int j = 0; j < 2; j++) acc[i][j] = vzero;

  int bk = tid >> 3, bn = (tid & 7) * 16;
#if !USE_TDM
  int arow = tid >> 1, aseg = (tid & 1) * 16;
#endif

  for (int k0 = 0; k0 < K; k0 += 32) {
#if USE_TDM
    // ---- stage A tile via Tensor Data Mover (wave 0 issues one 2D D#) ----
    if (tid < 32) {
      unsigned long long ga64 =
          (unsigned long long)(uintptr_t)A + (((size_t)m0 * K + k0) * 2ull);
      unsigned ldsA = (unsigned)(uintptr_t)&lA[0][0];
      u32x4 g0;
      g0[0] = 1u;                                       // count=1, user mode
      g0[1] = ldsA;                                     // lds_addr (bytes)
      g0[2] = (unsigned)(ga64 & 0xFFFFFFFFu);           // global_addr[31:0]
      g0[3] = (unsigned)((ga64 >> 32) & 0x01FFFFFFu)    // global_addr[56:32]
              | 0x80000000u;                            // type=2 ("image")
      i32x8 g1;
      g1[0] = (int)(1u << 16);                          // data_size=2B, mask=0
      g1[1] = (int)((unsigned)(K & 0xFFFF) << 16);      // tensor_dim0[15:0]=K (x)
      g1[2] = (int)(((unsigned)K >> 16) | ((unsigned)(M & 0xFFFF) << 16)); // dim0 hi | dim1 lo
      g1[3] = (int)(((unsigned)M >> 16) | (32u << 16)); // dim1 hi | tile_dim0=32
      g1[4] = (int)128u;                                // tile_dim1=128, tile_dim2=0
      g1[5] = (int)(unsigned)K;                         // tensor_dim0_stride=K
      g1[6] = 0;                                        // stride hi | dim1_stride lo
      g1[7] = 0;                                        // dim1_stride hi
      i32x4 z4 = {0, 0, 0, 0};
#if __clang_major__ >= 23
      i32x8 z8 = {0, 0, 0, 0, 0, 0, 0, 0};
      __builtin_amdgcn_tensor_load_to_lds(g0, g1, z4, z4, z8, 0);
#else
      __builtin_amdgcn_tensor_load_to_lds(g0, g1, z4, z4, 0);
#endif
      __builtin_amdgcn_s_wait_tensorcnt(0);
    }
#else
    // ---- stage A tile (row-major) with vector loads ----
    const __bf16* ga = A + (size_t)(m0 + arow) * K + k0 + aseg;
    *(uint4*)&lA[arow][aseg]     = *(const uint4*)ga;
    *(uint4*)&lA[arow][aseg + 8] = *(const uint4*)(ga + 8);
    if (k0 + 32 < K) __builtin_prefetch(ga + 32, 0, 1);
#endif
    // ---- stage B tile transposed to [n][k] ----
    const __bf16* gb = Bm + (size_t)(k0 + bk) * N + n0 + bn;
    __bf16 tb[16];
    *(uint4*)&tb[0] = *(const uint4*)gb;
    *(uint4*)&tb[8] = *(const uint4*)(gb + 8);
#pragma unroll
    for (int i = 0; i < 16; i++) lB[bn + i][bk] = tb[i];
    if (k0 + 32 < K) __builtin_prefetch(gb + (size_t)32 * N, 0, 1);
    __syncthreads();

    // ---- fragments from LDS (two b128 loads per fragment) ----
    v16bf af[4], bfv[2];
#pragma unroll
    for (int i = 0; i < 4; i++) {
      int m = mBase + i * 16 + lm;
      union { v16bf v; uint4 u[2]; } fa;
      fa.u[0] = *(const uint4*)&lA[m][kb];
      fa.u[1] = *(const uint4*)&lA[m][16 + kb];
      af[i] = fa.v;
    }
#pragma unroll
    for (int j = 0; j < 2; j++) {
      int n = nBase + j * 16 + lm;
      union { v16bf v; uint4 u[2]; } fb;
      fb.u[0] = *(const uint4*)&lB[n][kb];
      fb.u[1] = *(const uint4*)&lB[n][16 + kb];
      bfv[j] = fb.v;
    }
#pragma unroll
    for (int i = 0; i < 4; i++)
#pragma unroll
      for (int j = 0; j < 2; j++)
        acc[i][j] = __builtin_amdgcn_wmma_f32_16x16x32_bf16(
            false, af[i], false, bfv[j], (short)0, acc[i][j], false, false);
    __syncthreads();   // all waves done reading tiles before next stage/TDM
  }

  // ---- epilogue: lane l holds rows lh*8 + r, col lm of each 16x16 tile ----
#pragma unroll
  for (int i = 0; i < 4; i++)
#pragma unroll
    for (int j = 0; j < 2; j++)
#pragma unroll
      for (int r = 0; r < 8; r++) {
        int row = m0 + mBase + i * 16 + lh * 8 + r;
        int col = n0 + nBase + j * 16 + lm;
        Cm[(size_t)row * N + col] = acc[i][j][r];
      }
}

// ---------------- WKV linear-attention scan (one block per head) ----------------
__global__ __launch_bounds__(256)
void k_wkv(const float* __restrict__ R, const float* __restrict__ Kt,
           const float* __restrict__ Vt, const float* __restrict__ decay,
           const float* __restrict__ faaaa, float* __restrict__ out,
           float* __restrict__ stateOut) {
  __shared__ float sr[HSZ], sk[HSZ], sv[HSZ], red[4][HSZ];
  int h = blockIdx.x, tid = threadIdx.x;
  int v = tid & 63, q = tid >> 6;          // thread owns k in [16q,16q+16) for column v
  float s[16], w[16], u[16];
#pragma unroll
  for (int i = 0; i < 16; i++) {
    int kidx = h * HSZ + q * 16 + i;
    w[i] = __expf(-__expf(decay[kidx]));
    u[i] = faaaa[kidx];
    s[i] = 0.f;
  }
  for (int t = 0; t < T_LEN; t++) {
    int base = (t * NHEAD + h) * HSZ;
    if (tid < 64)       sr[tid]        = R[base + tid];
    else if (tid < 128) sk[tid - 64]   = Kt[base + tid - 64];
    else if (tid < 192) sv[tid - 128]  = Vt[base + tid - 128];
    __syncthreads();
    float vv = sv[v];
    float acc = 0.f;
#pragma unroll
    for (int i = 0; i < 16; i++) {
      float a = sk[q * 16 + i] * vv;
      acc = fmaf(sr[q * 16 + i], fmaf(u[i], a, s[i]), acc);   // r*(s_old + u*a)
      s[i] = fmaf(w[i], s[i], a);                             // s = w*s + a
    }
    red[q][v] = acc;
    __syncthreads();
    if (q == 0) out[base + v] = red[0][v] + red[1][v] + red[2][v] + red[3][v];
  }
#pragma unroll
  for (int i = 0; i < 16; i++)
    stateOut[(h * HSZ + q * 16 + i) * HSZ + v] = s[i];
}

// ---------------- GroupNorm(out/8)*w+b then * silu(g) -> bf16 ----------------
__global__ void k_gn_gate(const float* __restrict__ wkv, const float* __restrict__ g,
                          const float* __restrict__ gw, const float* __restrict__ gb,
                          __bf16* __restrict__ act) {
  __shared__ float red[HSZ];
  int th = blockIdx.x;          // t*NHEAD + h
  int hs = threadIdx.x;
  int h = th & (NHEAD - 1), t = th >> 4;
  float x = wkv[(size_t)th * HSZ + hs] * 0.125f;
  red[hs] = x; __syncthreads();
  for (int st = 32; st; st >>= 1) { if (hs < st) red[hs] += red[hs + st]; __syncthreads(); }
  float m = red[0] * (1.f / HSZ);
  __syncthreads();
  float d = x - m;
  red[hs] = d * d; __syncthreads();
  for (int st = 32; st; st >>= 1) { if (hs < st) red[hs] += red[hs + st]; __syncthreads(); }
  float rstd = rsqrtf(red[0] * (1.f / HSZ) + 1e-5f);
  int c = h * HSZ + hs;
  float val = d * rstd * gw[c] + gb[c];
  float gg = g[(size_t)t * C_DIM + c];
  float sil = gg / (1.f + __expf(-gg));
  act[(size_t)t * C_DIM + c] = (__bf16)(val * sil);
}

// ---------------- elementwise helpers ----------------
__global__ void k_add(float* __restrict__ x, const float* __restrict__ y, size_t n) {
  size_t i = (size_t)blockIdx.x * blockDim.x + threadIdx.x;
  size_t st = (size_t)gridDim.x * blockDim.x;
  for (; i < n; i += st) x[i] += y[i];
}

__global__ void k_relu_sq(const float* __restrict__ in, __bf16* __restrict__ out, size_t n) {
  size_t i = (size_t)blockIdx.x * blockDim.x + threadIdx.x;
  size_t st = (size_t)gridDim.x * blockDim.x;
  for (; i < n; i += st) { float r = fmaxf(in[i], 0.f); out[i] = (__bf16)(r * r); }
}

__global__ void k_sig_gate_add(float* __restrict__ x, const float* __restrict__ fr,
                               const float* __restrict__ kv, size_t n) {
  size_t i = (size_t)blockIdx.x * blockDim.x + threadIdx.x;
  size_t st = (size_t)gridDim.x * blockDim.x;
  for (; i < n; i += st) x[i] += kv[i] / (1.f + __expf(-fr[i]));
}

__global__ void k_copy_row(const float* __restrict__ src, float* __restrict__ dst) {
  for (int i = threadIdx.x; i < C_DIM; i += blockDim.x) dst[i] = src[i];
}

// ---------------- log-sum-exp + NLL per row ----------------
__global__ __launch_bounds__(256)
void k_lse_nll(const float* __restrict__ logits, const int* __restrict__ labels,
               float* __restrict__ nll, float* __restrict__ msk) {
  __shared__ float red[256];
  int t = blockIdx.x, tid = threadIdx.x;
  const float* row = logits + (size_t)t * VOCABN;
  float mx = -3.4e38f;
  for (int c = tid; c < VOCABN; c += 256) mx = fmaxf(mx, row[c]);
  red[tid] = mx; __syncthreads();
  for (int st = 128; st; st >>= 1) { if (tid < st) red[tid] = fmaxf(red[tid], red[tid + st]); __syncthreads(); }
  mx = red[0];
  __syncthreads();
  float se = 0.f;
  for (int c = tid; c < VOCABN; c += 256) se += __expf(row[c] - mx);
  red[tid] = se; __syncthreads();
  for (int st = 128; st; st >>= 1) { if (tid < st) red[tid] += red[tid + st]; __syncthreads(); }
  if (tid == 0) {
    float lse = mx + __logf(red[0]);
    int lbl = (t < T_LEN - 1) ? labels[t + 1] : -100;
    bool valid = (lbl >= 0) && (lbl < VOCABN);
    nll[t] = valid ? (lse - row[lbl]) : 0.f;
    msk[t] = valid ? 1.f : 0.f;
  }
}

__global__ void k_loss(const float* __restrict__ nll, const float* __restrict__ msk,
                       float* __restrict__ out0) {
  __shared__ float r1[256], r2[256];
  int tid = threadIdx.x;
  float s = 0.f, c = 0.f;
  for (int t = tid; t < T_LEN; t += 256) { s += nll[t]; c += msk[t]; }
  r1[tid] = s; r2[tid] = c; __syncthreads();
  for (int st = 128; st; st >>= 1) {
    if (tid < st) { r1[tid] += r1[tid + st]; r2[tid] += r2[tid + st]; }
    __syncthreads();
  }
  if (tid == 0) out0[0] = r1[0] / fmaxf(r2[0], 1.f);
}

// ------------------------------------------------------------------
// Host orchestration
// ------------------------------------------------------------------
struct LayerP {
  const float *Fk, *Fr, *Fv, *Wg, *Wk, *Wo, *Wr, *Wv;
  const float *cm_k, *cm_r;
  const float *ln0_b, *ln0_w;
  const float *ln1_b, *ln1_w, *ln2_b, *ln2_w, *lnx_b, *lnx_w;
  const float *td, *tf;
  const float *tm_g, *tm_k, *tm_r, *tm_v;
};

extern "C" void kernel_launch(void* const* d_in, const int* in_sizes, int n_in,
                              void* d_out, int out_size, void* d_ws, size_t ws_size,
                              hipStream_t stream) {
  (void)in_sizes; (void)n_in; (void)out_size; (void)ws_size;
  // Flattening assumption: JAX pytree (dict keys sorted alphabetically).
  // top: input_ids, labels, params{emb, head, layers[...], ln_out_b, ln_out_w}
  const int* ids    = (const int*)d_in[0];
  const int* labels = (const int*)d_in[1];
  const float* emb  = (const float*)d_in[2];
  const float* head = (const float*)d_in[3];
  LayerP L[NLAYER];
  int p = 4;
  for (int i = 0; i < NLAYER; i++) {
    L[i].Fk   = (const float*)d_in[p++];
    L[i].Fr   = (const float*)d_in[p++];
    L[i].Fv   = (const float*)d_in[p++];
    L[i].Wg   = (const float*)d_in[p++];
    L[i].Wk   = (const float*)d_in[p++];
    L[i].Wo   = (const float*)d_in[p++];
    L[i].Wr   = (const float*)d_in[p++];
    L[i].Wv   = (const float*)d_in[p++];
    L[i].cm_k = (const float*)d_in[p++];
    L[i].cm_r = (const float*)d_in[p++];
    if (i == 0) {
      L[i].ln0_b = (const float*)d_in[p++];
      L[i].ln0_w = (const float*)d_in[p++];
    } else { L[i].ln0_b = nullptr; L[i].ln0_w = nullptr; }
    L[i].ln1_b = (const float*)d_in[p++];
    L[i].ln1_w = (const float*)d_in[p++];
    L[i].ln2_b = (const float*)d_in[p++];
    L[i].ln2_w = (const float*)d_in[p++];
    L[i].lnx_b = (const float*)d_in[p++];
    L[i].lnx_w = (const float*)d_in[p++];
    L[i].td    = (const float*)d_in[p++];
    L[i].tf    = (const float*)d_in[p++];
    L[i].tm_g  = (const float*)d_in[p++];
    L[i].tm_k  = (const float*)d_in[p++];
    L[i].tm_r  = (const float*)d_in[p++];
    L[i].tm_v  = (const float*)d_in[p++];
  }
  const float* ln_out_b = (const float*)d_in[p++];
  const float* ln_out_w = (const float*)d_in[p++];

  // ---- workspace carve ----
  const size_t TC = (size_t)T_LEN * C_DIM;
  const size_t TF = (size_t)T_LEN * FFN;
  char* wsb = (char*)d_ws;
  size_t off = 0;
  auto take = [&](size_t bytes) -> void* {
    void* q = wsb + off;
    off += (bytes + 255) & ~(size_t)255;
    return q;
  };
  float* x    = (float*)take(TC * 4);
  float* ln   = (float*)take(TC * 4);
  float* rb   = (float*)take(TC * 4);
  float* kb   = (float*)take(TC * 4);
  float* vb   = (float*)take(TC * 4);
  float* gbuf = (float*)take(TC * 4);
  float* kvb  = (float*)take(TC * 4);
  float* frb  = (float*)take(TC * 4);
  float* wkvo = (float*)take(TC * 4);
  float* tmp  = (float*)take(TF * 4);
  __bf16* Abf = (__bf16*)take(TF * 2);
  __bf16* Wbf = (__bf16*)take((size_t)C_DIM * VOCABN * 2);
  float* nll  = (float*)take(T_LEN * 4);
  float* msk  = (float*)take(T_LEN * 4);

  // ---- output layout: loss | logits | states | shifts ----
  float* out    = (float*)d_out;
  float* logits = out + 1;
  float* states = logits + (size_t)T_LEN * VOCABN;
  float* shifts = states + (size_t)NLAYER * NHEAD * HSZ * HSZ;

  auto cvt = [&](const float* src, __bf16* dst, size_t n) {
    k_cvt<<<2048, 256, 0, stream>>>(src, dst, n);
  };
  auto gemm = [&](const __bf16* A, const __bf16* B, float* Cm, int M, int N, int K) {
    dim3 g(N / 128, M / 128);
    k_gemm_bf16<<<g, 256, 0, stream>>>(A, B, Cm, M, N, K);
  };

  // ---- embedding ----
  k_gather<<<T_LEN, 256, 0, stream>>>(emb, ids, x);

  for (int i = 0; i < NLAYER; i++) {
    const LayerP& lp = L[i];
    if (i == 0)
      k_layernorm<<<T_LEN, 256, 0, stream>>>(x, x, lp.ln0_w, lp.ln0_b);

    // ---- time mix ----
    k_layernorm<<<T_LEN, 256, 0, stream>>>(x, ln, lp.ln1_w, lp.ln1_b);
    k_copy_row<<<1, 256, 0, stream>>>(ln + (size_t)(T_LEN - 1) * C_DIM,
                                      shifts + (size_t)(i * 2 + 0) * C_DIM);
    k_mix<<<T_LEN, 256, 0, stream>>>(ln, lp.tm_r, Abf);
    cvt(lp.Wr, Wbf, (size_t)C_DIM * C_DIM);
    gemm(Abf, Wbf, rb, T_LEN, C_DIM, C_DIM);
    k_mix<<<T_LEN, 256, 0, stream>>>(ln, lp.tm_k, Abf);
    cvt(lp.Wk, Wbf, (size_t)C_DIM * C_DIM);
    gemm(Abf, Wbf, kb, T_LEN, C_DIM, C_DIM);
    k_mix<<<T_LEN, 256, 0, stream>>>(ln, lp.tm_v, Abf);
    cvt(lp.Wv, Wbf, (size_t)C_DIM * C_DIM);
    gemm(Abf, Wbf, vb, T_LEN, C_DIM, C_DIM);
    k_mix<<<T_LEN, 256, 0, stream>>>(ln, lp.tm_g, Abf);
    cvt(lp.Wg, Wbf, (size_t)C_DIM * C_DIM);
    gemm(Abf, Wbf, gbuf, T_LEN, C_DIM, C_DIM);

    k_wkv<<<NHEAD, 256, 0, stream>>>(rb, kb, vb, lp.td, lp.tf, wkvo,
                                     states + (size_t)i * NHEAD * HSZ * HSZ);
    k_gn_gate<<<T_LEN * NHEAD, HSZ, 0, stream>>>(wkvo, gbuf, lp.lnx_w, lp.lnx_b, Abf);
    cvt(lp.Wo, Wbf, (size_t)C_DIM * C_DIM);
    gemm(Abf, Wbf, tmp, T_LEN, C_DIM, C_DIM);
    k_add<<<1024, 256, 0, stream>>>(x, tmp, TC);

    // ---- channel mix ----
    k_layernorm<<<T_LEN, 256, 0, stream>>>(x, ln, lp.ln2_w, lp.ln2_b);
    k_copy_row<<<1, 256, 0, stream>>>(ln + (size_t)(T_LEN - 1) * C_DIM,
                                      shifts + (size_t)(i * 2 + 1) * C_DIM);
    k_mix<<<T_LEN, 256, 0, stream>>>(ln, lp.cm_k, Abf);
    cvt(lp.Fk, Wbf, (size_t)C_DIM * FFN);
    gemm(Abf, Wbf, tmp, T_LEN, FFN, C_DIM);
    k_relu_sq<<<2048, 256, 0, stream>>>(tmp, Abf, TF);
    cvt(lp.Fv, Wbf, (size_t)FFN * C_DIM);
    gemm(Abf, Wbf, kvb, T_LEN, C_DIM, FFN);
    k_mix<<<T_LEN, 256, 0, stream>>>(ln, lp.cm_r, Abf);
    cvt(lp.Fr, Wbf, (size_t)C_DIM * C_DIM);
    gemm(Abf, Wbf, frb, T_LEN, C_DIM, C_DIM);
    k_sig_gate_add<<<1024, 256, 0, stream>>>(x, frb, kvb, TC);
  }

  // ---- head + loss ----
  k_layernorm<<<T_LEN, 256, 0, stream>>>(x, ln, ln_out_w, ln_out_b);
  cvt(ln, Abf, TC);
  cvt(head, Wbf, (size_t)C_DIM * VOCABN);
  gemm(Abf, Wbf, logits, T_LEN, VOCABN, C_DIM);
  k_lse_nll<<<T_LEN, 256, 0, stream>>>(logits, labels, nll, msk);
  k_loss<<<1, 256, 0, stream>>>(nll, msk, out);
}
